// SelfAttention_31808527794483
// MI455X (gfx1250) — compile-verified, matrix operand order checked by reading, not run on hardware
//
#include <hip/hip_runtime.h>

typedef __attribute__((ext_vector_type(16))) _Float16 v16h;
typedef __attribute__((ext_vector_type(8)))  _Float16 v8h;
typedef __attribute__((ext_vector_type(8)))  float    v8f;

#define DEV __device__ __forceinline__

static constexpr int B_  = 512;
static constexpr int T_  = 256;
static constexpr int D_  = 384;
static constexpr int H_  = 64;
static constexpr int BT_ = B_ * T_;

DEV v8f wmma_f32_16x16x32(v16h a, v16h b, v8f c) {
  // (neg_a, A, neg_b, B, c_mod, C, reuse_a, reuse_b)
  return __builtin_amdgcn_wmma_f32_16x16x32_f16(false, a, false, b, (short)0, c,
                                                false, false);
}

// A fragment (16x32, f16), element (m,k) at p[m*ld + k] (row-major).
// ISA layout: lane = half*16 + m; per lane two contiguous 8-element K runs.
DEV v16h load_a_f16(const _Float16* p, int ld, int row0, int col0, int lane) {
  int hf = (lane >> 4) & 1;
  int m  = lane & 15;
  const _Float16* b0 = p + (size_t)(row0 + m) * ld + col0 + hf * 8;
  v8h lo = *(const v8h*)(b0);
  v8h hi = *(const v8h*)(b0 + 16);
  v16h a;
#pragma unroll
  for (int i = 0; i < 8; ++i) { a[i] = lo[i]; a[i + 8] = hi[i]; }
  return a;
}

// A fragment from f32 memory with on-the-fly f16 conversion.
DEV v16h load_a_f32(const float* p, int ld, int row0, int col0, int lane) {
  int hf = (lane >> 4) & 1;
  int m  = lane & 15;
  const float* b0 = p + (size_t)(row0 + m) * ld + col0 + hf * 8;
  v16h a;
#pragma unroll
  for (int i = 0; i < 8; ++i) {
    a[i]     = (_Float16)b0[i];
    a[i + 8] = (_Float16)b0[i + 16];
  }
  return a;
}

// B fragment (32x16, f16), element (k,n) stored K-contiguous: p[n*ldn + k].
DEV v16h load_b_kmaj(const _Float16* p, int ldn, int n0, int k0, int lane) {
  int hf = (lane >> 4) & 1;
  int n  = lane & 15;
  const _Float16* b0 = p + (size_t)(n0 + n) * ldn + k0 + hf * 8;
  v8h lo = *(const v8h*)(b0);
  v8h hi = *(const v8h*)(b0 + 16);
  v16h b;
#pragma unroll
  for (int i = 0; i < 8; ++i) { b[i] = lo[i]; b[i + 8] = hi[i]; }
  return b;
}

// ---------------------------------------------------------------------------
// Kernel 0: transpose the three D x H f32 weight matrices into f16
// Wt[proj][h][d] (144 KB -> permanently L2-resident). The fused QKV kernel
// then stages this into LDS with a straight contiguous b128 copy.
// ---------------------------------------------------------------------------
__global__ __launch_bounds__(256) void wt_kernel(
    const float* __restrict__ Wq, const float* __restrict__ Wk,
    const float* __restrict__ Wv, _Float16* __restrict__ wt) {
  const float* W = (blockIdx.x == 0) ? Wq : (blockIdx.x == 1) ? Wk : Wv;
  _Float16* dst = wt + (size_t)blockIdx.x * H_ * D_;
  for (int idx = threadIdx.x; idx < D_ * H_; idx += blockDim.x) {
    int d = idx >> 6;  // H_ == 64
    int h = idx & 63;
    dst[h * D_ + d] = (_Float16)W[idx];
  }
}

// ---------------------------------------------------------------------------
// Kernel 1: fused QKV projection, ONE pass over x (the dominant HBM traffic).
// All three transposed weight matrices live in LDS (144 KB of the WGP's
// 320 KB) so B-fragments are low-latency ds_load_b128 the scheduler can
// pipeline against WMMAs (instead of serialized L2 round-trips).
// grid = BT/256, block = 256 (8 waves); each wave computes two 16-row tiles
// of Q, K and V: per K-step one x A-fragment feeds 12 WMMAs.
// Q/K stored row-major f16 [BT,64]; V stored transposed per batch [64,256].
// ---------------------------------------------------------------------------
__global__ __launch_bounds__(256) void qkv_fused_kernel(
    const float* __restrict__ x,
    const float* __restrict__ bq, const float* __restrict__ bk,
    const float* __restrict__ bv, const _Float16* __restrict__ wt,
    _Float16* __restrict__ qbuf, _Float16* __restrict__ kbuf,
    _Float16* __restrict__ vtbuf) {
  __shared__ __align__(16) _Float16 Wt[3 * H_ * D_];  // 144 KB

  // Stage all three transposed weights: contiguous b128 global -> b128 ds.
  for (int idx = threadIdx.x * 8; idx < 3 * H_ * D_; idx += blockDim.x * 8) {
    *(v8h*)(Wt + idx) = *(const v8h*)(wt + idx);
  }
  __syncthreads();

  const int wave = threadIdx.x >> 5;
  const int lane = threadIdx.x & 31;
  const int hf   = (lane >> 4) & 1;
  const int lq   = lane & 15;

  for (int sub = 0; sub < 2; ++sub) {
    const int row0 = (blockIdx.x * 16 + wave * 2 + sub) * 16;  // in [0, BT)

    v8f acc[3][4] = {};
    for (int kk = 0; kk < 12; ++kk) {
      v16h a = load_a_f32(x, D_, row0, kk * 32, lane);
#pragma unroll
      for (int p = 0; p < 3; ++p) {
#pragma unroll
        for (int ht = 0; ht < 4; ++ht) {
          v16h bw = load_b_kmaj(Wt + p * H_ * D_, D_, ht * 16, kk * 32, lane);
          acc[p][ht] = wmma_f32_16x16x32(a, bw, acc[p][ht]);
        }
      }
    }

    // Q (p=0) and K (p=1): row-major f16.
#pragma unroll
    for (int p = 0; p < 2; ++p) {
      _Float16* out = (p == 0) ? qbuf : kbuf;
      const float* bias = (p == 0) ? bq : bk;
#pragma unroll
      for (int ht = 0; ht < 4; ++ht) {
        float bb = bias[ht * 16 + lq];
#pragma unroll
        for (int r = 0; r < 8; ++r) {
          int row = row0 + r + 8 * hf;
          out[(size_t)row * H_ + ht * 16 + lq] = (_Float16)(acc[p][ht][r] + bb);
        }
      }
    }
    // V: store transposed per batch, Vt[b][h][t] (contiguous v8h per lane).
    {
      int bidx = row0 / T_;
      int tb   = row0 % T_;
#pragma unroll
      for (int ht = 0; ht < 4; ++ht) {
        float bb = bv[ht * 16 + lq];
        v8h pk;
#pragma unroll
        for (int r = 0; r < 8; ++r) pk[r] = (_Float16)(acc[2][ht][r] + bb);
        _Float16* dst = vtbuf + (size_t)bidx * H_ * T_ +
                        (size_t)(ht * 16 + lq) * T_ + tb + hf * 8;
        *(v8h*)dst = pk;
      }
    }
  }
}

// ---------------------------------------------------------------------------
// Kernel 2: fused causal attention (flash style). One block (8 waves) per
// batch. Causal-balanced mapping: wave w owns q-tiles {w, 15-w} so every
// wave runs 9 key-tile iterations (instead of 2..16). Per 32-col key
// tile: 4 WMMAs for scores, online softmax in registers (shfl_xor row
// reductions within 16-lane halves), P round-trips through 1KB/wave LDS to
// convert D-layout -> A-layout, then 4 WMMAs accumulate P @ V.
// ---------------------------------------------------------------------------
__global__ __launch_bounds__(256) void attn_kernel(
    const _Float16* __restrict__ qbuf, const _Float16* __restrict__ kbuf,
    const _Float16* __restrict__ vtbuf, float* __restrict__ out) {
  constexpr float SCALE = 0.0625f;  // CONTEXT_SIZE^-0.5 = 1/16
  __shared__ __align__(16) _Float16 pbuf[8][16 * 32];

  const int b = blockIdx.x;
  const _Float16* Q  = qbuf  + (size_t)b * T_ * H_;
  const _Float16* K  = kbuf  + (size_t)b * T_ * H_;
  const _Float16* Vt = vtbuf + (size_t)b * H_ * T_;
  float* O = out + (size_t)b * T_ * H_;

  const int wave = threadIdx.x >> 5;
  const int lane = threadIdx.x & 31;
  const int hf   = (lane >> 4) & 1;
  const int lq   = lane & 15;
  _Float16* pw = &pbuf[wave][0];

  for (int qt = 0; qt < 2; ++qt) {
    const int qtile = (qt == 0) ? wave : (15 - wave);  // balanced causal work
    const int qb    = qtile * 16;                      // query tile start row
    v16h a0 = load_a_f16(Q, H_, qb, 0, lane);
    v16h a1 = load_a_f16(Q, H_, qb, 32, lane);

    v8f acc[4] = {};
    float rm[8], rl[8];
#pragma unroll
    for (int r = 0; r < 8; ++r) { rm[r] = -__builtin_inff(); rl[r] = 0.0f; }

    const int nj = qb / 32 + 1;  // causal: key tiles with sb <= qb+15
    for (int j = 0; j < nj; ++j) {
      const int sb = j * 32;
      // Scores S[16 x 32] = Q(16x64) @ K^T, two 16x16 D tiles, 2 K-steps.
      v16h bk00 = load_b_kmaj(K, H_, sb,      0,  lane);
      v16h bk01 = load_b_kmaj(K, H_, sb,      32, lane);
      v16h bk10 = load_b_kmaj(K, H_, sb + 16, 0,  lane);
      v16h bk11 = load_b_kmaj(K, H_, sb + 16, 32, lane);
      v8f s0 = {}, s1 = {};
      s0 = wmma_f32_16x16x32(a0, bk00, s0);
      s0 = wmma_f32_16x16x32(a1, bk01, s0);
      s1 = wmma_f32_16x16x32(a0, bk10, s1);
      s1 = wmma_f32_16x16x32(a1, bk11, s1);

#pragma unroll
      for (int r = 0; r < 8; ++r) {
        const int m  = qb + r + 8 * hf;       // global query row of this elem
        const int n0 = sb + lq;
        const int n1 = sb + 16 + lq;
        float v0 = (n0 <= m) ? s0[r] * SCALE : -__builtin_inff();
        float v1 = (n1 <= m) ? s1[r] * SCALE : -__builtin_inff();
        // Row max across the 16 lanes sharing this row (xor of low 4 bits
        // stays within each half-wave).
        float t = fmaxf(v0, v1);
        t = fmaxf(t, __shfl_xor(t, 1));
        t = fmaxf(t, __shfl_xor(t, 2));
        t = fmaxf(t, __shfl_xor(t, 4));
        t = fmaxf(t, __shfl_xor(t, 8));
        float newm = fmaxf(rm[r], t);
        float corr = __expf(rm[r] - newm);
        float p0 = __expf(v0 - newm);
        float p1 = __expf(v1 - newm);
        float rs = p0 + p1;
        rs += __shfl_xor(rs, 1);
        rs += __shfl_xor(rs, 2);
        rs += __shfl_xor(rs, 4);
        rs += __shfl_xor(rs, 8);
        rl[r] = rl[r] * corr + rs;
        rm[r] = newm;
#pragma unroll
        for (int ht = 0; ht < 4; ++ht) acc[ht][r] *= corr;
        // Stage P row-major into this wave's LDS tile (D-layout -> A-layout).
        const int mrow = r + 8 * hf;
        pw[mrow * 32 + lq]      = (_Float16)p0;
        pw[mrow * 32 + 16 + lq] = (_Float16)p1;
      }
      // Same-wave LDS ops are in order: no barrier needed (per-wave buffer).
      v16h ap = load_a_f16(pw, 32, 0, 0, lane);
#pragma unroll
      for (int ht = 0; ht < 4; ++ht) {
        v16h bv = load_b_kmaj(Vt, T_, ht * 16, sb, lane);
        acc[ht] = wmma_f32_16x16x32(ap, bv, acc[ht]);
      }
    }

    // Epilogue: normalize by row sums and store f32 output.
#pragma unroll
    for (int ht = 0; ht < 4; ++ht) {
#pragma unroll
      for (int r = 0; r < 8; ++r) {
        int row = qb + r + 8 * hf;
        O[(size_t)row * H_ + ht * 16 + lq] = acc[ht][r] / rl[r];
      }
    }
  }
}

extern "C" void kernel_launch(void* const* d_in, const int* in_sizes, int n_in,
                              void* d_out, int out_size, void* d_ws,
                              size_t ws_size, hipStream_t stream) {
  (void)in_sizes; (void)n_in; (void)out_size; (void)ws_size;
  const float* x  = (const float*)d_in[0];
  const float* Wq = (const float*)d_in[1];
  const float* bq = (const float*)d_in[2];
  const float* Wk = (const float*)d_in[3];
  const float* bk = (const float*)d_in[4];
  const float* Wv = (const float*)d_in[5];
  const float* bv = (const float*)d_in[6];
  float* out = (float*)d_out;

  _Float16* qbuf  = (_Float16*)d_ws;                 // [BT, 64] f16
  _Float16* kbuf  = qbuf + (size_t)BT_ * H_;         // [BT, 64] f16
  _Float16* vtbuf = kbuf + (size_t)BT_ * H_;         // [B][64][256] f16
  _Float16* wtbuf = vtbuf + (size_t)B_ * H_ * T_;    // [3][64][384] f16

  wt_kernel<<<3, 256, 0, stream>>>(Wq, Wk, Wv, wtbuf);
  qkv_fused_kernel<<<BT_ / 256, 256, 0, stream>>>(x, bq, bk, bv, wtbuf,
                                                  qbuf, kbuf, vtbuf);
  attn_kernel<<<B_, 256, 0, stream>>>(qbuf, kbuf, vtbuf, out);
}
